// MultiheadAttention_47459388621498
// MI455X (gfx1250) — compile-verified, matrix operand order checked by reading, not run on hardware
//
#include <hip/hip_runtime.h>
#include <hip/hip_bf16.h>

// ---------------------------------------------------------------------------
// MultiheadAttention (cross attention) for MI455X / gfx1250, wave32 + WMMA.
// D=768, H=12, DH=64, b=2, n1=n2=4096.  All matmuls use
// V_WMMA_F32_16X16X32_F16 in NT form (both operands loaded with the same
// row-major 16x32 fragment loader => D = X * Y^T).
// Attention stages K / V^T chunks into double-buffered LDS with
// GLOBAL_LOAD_ASYNC_TO_LDS_B128 (ASYNCcnt) when the builtin is available.
// ---------------------------------------------------------------------------

typedef __attribute__((ext_vector_type(16))) _Float16 v16h;
typedef __attribute__((ext_vector_type(8)))  _Float16 v8h;
typedef __attribute__((ext_vector_type(8)))  float    v8f;
typedef int v4i __attribute__((vector_size(16)));   // 128-bit payload type

#define D_MODEL 768
#define NH      12
#define DH      64
#define SEQ     4096
#define ROWS    8192   // b * seq

#if defined(__has_builtin)
#if __has_builtin(__builtin_amdgcn_global_load_async_to_lds_b128) && \
    __has_builtin(__builtin_amdgcn_s_wait_asynccnt)
#define USE_ASYNC_LDS 1
#endif
#endif

#ifdef USE_ASYNC_LDS
#define ASYNC_WAIT(n) __builtin_amdgcn_s_wait_asynccnt(n)
#else
#define ASYNC_WAIT(n)
#endif

// Pointer-to-v4i in explicit address spaces (AS1 = __device__, AS3 = __shared__).
typedef __attribute__((address_space(1))) v4i* gas1_v4i;
typedef __attribute__((address_space(3))) v4i* las3_v4i;

// 16-byte global -> LDS copy; async DMA path on gfx1250 toolchains that
// expose the builtin, register relay otherwise.
__device__ __forceinline__ void copy16_g2l(const _Float16* g, _Float16* l) {
#ifdef USE_ASYNC_LDS
    __builtin_amdgcn_global_load_async_to_lds_b128(
        (gas1_v4i)(uintptr_t)g,
        (las3_v4i)(uint32_t)(uintptr_t)l,
        0, 0);
#else
    *reinterpret_cast<v8h*>(l) = *reinterpret_cast<const v8h*>(g);
#endif
}

union Frag16 { v16h v; v8h h[2]; };

// Load a 16x32 f16 tile (row-major, row stride ld halves) in the A/B-operand
// lane layout of V_WMMA_F32_16X16X32_F16:
//   lane l: row m = l&15, half = l>>4; VGPR0-3 hold K = half*8 .. half*8+7,
//   VGPR4-7 hold K = 16 + half*8 .. 16 + half*8 + 7.
__device__ __forceinline__ v16h load_frag(const _Float16* __restrict__ base, int ld) {
    const int lane = threadIdx.x & 31;
    const int m    = lane & 15;
    const int half = lane >> 4;
    const _Float16* p = base + (size_t)m * ld + half * 8;
    Frag16 f;
    f.h[0] = *reinterpret_cast<const v8h*>(p);
    f.h[1] = *reinterpret_cast<const v8h*>(p + 16);
    return f.v;
}

__device__ __forceinline__ v8f wmma_f16(v16h a, v16h b, v8f c) {
    return __builtin_amdgcn_wmma_f32_16x16x32_f16(
        /*neg_a=*/false, a, /*neg_b=*/false, b,
        /*c_mod=*/(short)0, c, /*reuse_a=*/false, /*reuse_b=*/false);
}

// ---------------------------------------------------------------------------
// LayerNorm (f32 in) -> f16 out, one block per row of 768.
// ---------------------------------------------------------------------------
__global__ __launch_bounds__(256) void ln_f16_kernel(
    const float* __restrict__ x, const float* __restrict__ g,
    const float* __restrict__ bta, _Float16* __restrict__ out) {
    const int row = blockIdx.x;
    const float* xr = x + (size_t)row * D_MODEL;
    const int t = threadIdx.x;
    float v[3];
    float s = 0.f, ss = 0.f;
#pragma unroll
    for (int i = 0; i < 3; ++i) {
        float val = xr[t + 256 * i];
        v[i] = val; s += val; ss += val * val;
    }
#pragma unroll
    for (int m = 1; m < 32; m <<= 1) {
        s  += __shfl_xor(s,  m, 32);
        ss += __shfl_xor(ss, m, 32);
    }
    __shared__ float red_s[8], red_ss[8];
    if ((t & 31) == 0) { red_s[t >> 5] = s; red_ss[t >> 5] = ss; }
    __syncthreads();
    s = 0.f; ss = 0.f;
#pragma unroll
    for (int i = 0; i < 8; ++i) { s += red_s[i]; ss += red_ss[i]; }
    const float mu   = s  * (1.f / D_MODEL);
    const float var  = ss * (1.f / D_MODEL) - mu * mu;
    const float rinv = rsqrtf(var + 1e-5f);
    _Float16* orow = out + (size_t)row * D_MODEL;
#pragma unroll
    for (int i = 0; i < 3; ++i) {
        const int c = t + 256 * i;
        orow[c] = (_Float16)((v[i] - mu) * rinv * g[c] + bta[c]);
    }
}

// ---------------------------------------------------------------------------
// f32 -> f16 weight conversion.
// ---------------------------------------------------------------------------
__global__ void cvt_f32_f16(const float* __restrict__ in,
                            _Float16* __restrict__ out, int n) {
    const int i = blockIdx.x * blockDim.x + threadIdx.x;
    if (i < n) out[i] = (_Float16)in[i];
}

// ---------------------------------------------------------------------------
// NT GEMM: C[ROWS x 768] = A[ROWS x 768] * W^T + bias.
// Fully unrolled k-loop, unroll-by-2 ping-pong double buffering: loads for
// chunk kc+32 issue before the WMMAs of chunk kc; the 'if' guards fold away
// at compile time, so no register rotation movs and no WAR hazard nops.
//   mode 0: write f16 to [b, h, n, 64]     (Q / K head layout)
//   mode 1: write f16 to [b, h, 64, n]     (V transposed head layout)
//   mode 2: write f32 to d_out with bias + residual (output projection)
// ---------------------------------------------------------------------------
__global__ __launch_bounds__(256) void gemm_nt_kernel(
    const _Float16* __restrict__ A, const _Float16* __restrict__ W,
    const float* __restrict__ bias, void* __restrict__ outp,
    const float* __restrict__ residual, int mode) {
    const int wave = threadIdx.x >> 5;
    const int lane = threadIdx.x & 31;
    const int n    = lane & 15;
    const int half = lane >> 4;
    const int Rm   = (blockIdx.x * 8 + wave) * 16;
    const int Nb   = blockIdx.y * 64;

    const _Float16* arow = A + (size_t)Rm * D_MODEL;
    const _Float16* wrow = W + (size_t)Nb * D_MODEL;

    v8f acc[4] = {};
    // prologue: fragments for kc = 0
    v16h af0 = load_frag(arow, D_MODEL);
    v16h bf0[4];
#pragma unroll
    for (int tt = 0; tt < 4; ++tt)
        bf0[tt] = load_frag(wrow + (size_t)tt * 16 * D_MODEL, D_MODEL);

#pragma unroll
    for (int kc = 0; kc < D_MODEL; kc += 64) {
        // stage 1 loads (kc+32 always < 768 since 768/32 is even)
        v16h af1 = load_frag(arow + kc + 32, D_MODEL);
        v16h bf1[4];
#pragma unroll
        for (int tt = 0; tt < 4; ++tt)
            bf1[tt] = load_frag(wrow + (size_t)tt * 16 * D_MODEL + kc + 32, D_MODEL);
        // stage 0 WMMAs
#pragma unroll
        for (int tt = 0; tt < 4; ++tt)
            acc[tt] = wmma_f16(af0, bf0[tt], acc[tt]);
        // stage 0 loads for next unrolled iteration (statically resolved)
        if (kc + 64 < D_MODEL) {
            af0 = load_frag(arow + kc + 64, D_MODEL);
#pragma unroll
            for (int tt = 0; tt < 4; ++tt)
                bf0[tt] = load_frag(wrow + (size_t)tt * 16 * D_MODEL + kc + 64, D_MODEL);
        }
        // stage 1 WMMAs
#pragma unroll
        for (int tt = 0; tt < 4; ++tt)
            acc[tt] = wmma_f16(af1, bf1[tt], acc[tt]);
    }

#pragma unroll
    for (int tt = 0; tt < 4; ++tt) {
        const int col = Nb + tt * 16 + n;      // fixed per lane: C layout lane=column
        const float bv = bias[col];
        if (mode == 0) {
            const int h = col >> 6, d = col & 63;
#pragma unroll
            for (int j = 0; j < 8; ++j) {      // VGPR j holds row Rm + half*8 + j
                const int r  = Rm + half * 8 + j;
                const int bb = r >> 12, m = r & 4095;
                ((_Float16*)outp)[((size_t)(bb * NH + h) * SEQ + m) * DH + d] =
                    (_Float16)(acc[tt][j] + bv);
            }
        } else if (mode == 1) {
            const int h = col >> 6, d = col & 63;
            const int r0 = Rm + half * 8;
            const int bb = r0 >> 12, m0 = r0 & 4095;
            v8h pk;
#pragma unroll
            for (int j = 0; j < 8; ++j) pk[j] = (_Float16)(acc[tt][j] + bv);
            *reinterpret_cast<v8h*>((_Float16*)outp +
                ((size_t)(bb * NH + h) * DH + d) * SEQ + m0) = pk;  // m contiguous
        } else {
            float* of = (float*)outp;
#pragma unroll
            for (int j = 0; j < 8; ++j) {
                const size_t idx = (size_t)(Rm + half * 8 + j) * D_MODEL + col;
                of[idx] = acc[tt][j] + bv + residual[idx];
            }
        }
    }
}

// ---------------------------------------------------------------------------
// Flash attention. Block = 8 waves sharing one (b,h); each wave owns a
// 16-query tile. K chunk (32x64 f16) and V^T chunk (64x32 f16) are staged
// into double-buffered LDS cooperatively (async DMA when available); online
// softmax in f32 with 16-lane shfl_xor row reductions; P goes through a
// per-wave LDS round trip (C-layout f32 -> A-layout f16).
// ---------------------------------------------------------------------------
__global__ __launch_bounds__(256) void attn_kernel(
    const _Float16* __restrict__ qb,   // [b,h,4096,64]
    const _Float16* __restrict__ kb,   // [b,h,4096,64]
    const _Float16* __restrict__ vt,   // [b,h,64,4096]
    _Float16* __restrict__ ctx) {      // [b,4096,768]
    __shared__ _Float16 k_s[2][32 * DH];   // 2 x 4 KB
    __shared__ _Float16 v_s[2][DH * 32];   // 2 x 4 KB
    __shared__ _Float16 p_s[8][16 * 32];   // 8 KB (per-wave P buffers)

    const int t    = threadIdx.x;
    const int wave = t >> 5;
    const int lane = t & 31;
    const int n    = lane & 15;
    const int half = lane >> 4;
    const int gw   = blockIdx.x * 8 + wave;   // 0..6143
    const int bh   = gw >> 8;                 // all 8 waves share bh
    const int q0   = (gw & 255) * 16;

    const _Float16* qbase = qb + ((size_t)bh * SEQ + q0) * DH;
    const _Float16* kbase = kb + (size_t)bh * SEQ * DH;
    const _Float16* vbase = vt + (size_t)bh * DH * SEQ;

    // staging assignment: 256 threads x 2 x 16B per chunk pair
    const int krow = t >> 3, kc8 = (t & 7) * 8;   // K: 32 rows x 64 halves
    const int vrow = t >> 2, vc8 = (t & 3) * 8;   // V^T: 64 rows x 32 halves

    const v16h qa0 = load_frag(qbase, DH);        // Q cols 0..31
    const v16h qa1 = load_frag(qbase + 32, DH);   // Q cols 32..63

    v8f acc[4] = {};
    float rm[8], rl[8];
#pragma unroll
    for (int j = 0; j < 8; ++j) { rm[j] = -1e30f; rl[j] = 0.f; }

    _Float16* pl = p_s[wave];

    // prologue: stage chunk 0 into buffer 0
    copy16_g2l(kbase + (size_t)krow * DH + kc8, k_s[0] + krow * DH + kc8);
    copy16_g2l(vbase + (size_t)vrow * SEQ + vc8, v_s[0] + vrow * 32 + vc8);

#pragma unroll 2
    for (int k0 = 0; k0 < SEQ; k0 += 32) {
        const int buf = (k0 >> 5) & 1;     // constant within each unrolled half
        if (k0 + 32 < SEQ) {               // stage next chunk into other buffer
            const int nb = buf ^ 1;
            copy16_g2l(kbase + (size_t)(k0 + 32 + krow) * DH + kc8,
                       k_s[nb] + krow * DH + kc8);
            copy16_g2l(vbase + (size_t)vrow * SEQ + (k0 + 32) + vc8,
                       v_s[nb] + vrow * 32 + vc8);
            ASYNC_WAIT(2);                 // in-order: current chunk has landed
        } else {
            ASYNC_WAIT(0);
        }
        __syncthreads();                   // all waves' staging visible

        const _Float16* ks = k_s[buf];
        const _Float16* vs = v_s[buf];

        // scores S[16 x 32] = Q(16x64) * K_chunk(32x64)^T, two 16x16 tiles
        v8f s0 = {}, s1 = {};
        {
            const v16h kf0 = load_frag(ks, DH);
            const v16h kf1 = load_frag(ks + 32, DH);
            s0 = wmma_f16(qa0, kf0, s0);
            s0 = wmma_f16(qa1, kf1, s0);
            const v16h kf2 = load_frag(ks + 16 * DH, DH);
            const v16h kf3 = load_frag(ks + 16 * DH + 32, DH);
            s1 = wmma_f16(qa0, kf2, s1);
            s1 = wmma_f16(qa1, kf3, s1);
        }

        // online softmax; C-layout: VGPR j <-> row half*8+j, lane n <-> column
        float p0[8], p1[8];
#pragma unroll
        for (int j = 0; j < 8; ++j) {
            const float a  = s0[j] * 0.125f;   // 1/sqrt(64)
            const float b2 = s1[j] * 0.125f;
            float mx = fmaxf(a, b2);
#pragma unroll
            for (int msk = 1; msk < 16; msk <<= 1)
                mx = fmaxf(mx, __shfl_xor(mx, msk, 32));  // within 16-lane half
            const float mnew  = fmaxf(rm[j], mx);
            const float alpha = __expf(rm[j] - mnew);
            p0[j] = __expf(a - mnew);
            p1[j] = __expf(b2 - mnew);
            float sum = p0[j] + p1[j];
#pragma unroll
            for (int msk = 1; msk < 16; msk <<= 1)
                sum += __shfl_xor(sum, msk, 32);
            rl[j] = rl[j] * alpha + sum;
            rm[j] = mnew;
            acc[0][j] *= alpha; acc[1][j] *= alpha;
            acc[2][j] *= alpha; acc[3][j] *= alpha;
        }

        // P (f32, C layout) -> LDS row-major f16 16x32 -> A fragment
#pragma unroll
        for (int j = 0; j < 8; ++j) {
            pl[(half * 8 + j) * 32 + n]      = (_Float16)p0[j];
            pl[(half * 8 + j) * 32 + 16 + n] = (_Float16)p1[j];
        }
        asm volatile("s_wait_dscnt 0" ::: "memory");
        const v16h pf = load_frag(pl, 32);

        // ctx += P(16x32) * V_chunk ; B fragments from V^T rows (NT form)
#pragma unroll
        for (int tt = 0; tt < 4; ++tt) {
            const v16h vf = load_frag(vs + tt * 16 * 32, 32);
            acc[tt] = wmma_f16(pf, vf, acc[tt]);
        }
        __syncthreads();                   // protect buffer reuse two stages later
    }

    const int b = bh / NH, h = bh % NH;
    _Float16* cbase = ctx + ((size_t)b * SEQ + q0) * D_MODEL + h * DH;
#pragma unroll
    for (int j = 0; j < 8; ++j) {
        const float inv = 1.f / rl[j];
#pragma unroll
        for (int tt = 0; tt < 4; ++tt)
            cbase[(size_t)(half * 8 + j) * D_MODEL + tt * 16 + n] =
                (_Float16)(acc[tt][j] * inv);
    }
}

// ---------------------------------------------------------------------------
extern "C" void kernel_launch(void* const* d_in, const int* in_sizes, int n_in,
                              void* d_out, int out_size, void* d_ws, size_t ws_size,
                              hipStream_t stream) {
    const float* input_cap  = (const float*)d_in[0];
    const float* output_cap = (const float*)d_in[1];
    const float* Wq = (const float*)d_in[2];
    const float* bq = (const float*)d_in[3];
    const float* Wk = (const float*)d_in[4];
    const float* bk = (const float*)d_in[5];
    const float* Wv = (const float*)d_in[6];
    const float* bv = (const float*)d_in[7];
    const float* ln_g = (const float*)d_in[8];
    const float* ln_b = (const float*)d_in[9];
    const float* Wp = (const float*)d_in[10];
    const float* bp = (const float*)d_in[11];

    // workspace carve-up (f16 elements); total ~80.3 MB
    _Float16* ws16 = (_Float16*)d_ws;
    size_t off = 0;
    const size_t NROW = (size_t)ROWS * D_MODEL;   // 6,291,456
    const size_t NW   = (size_t)D_MODEL * D_MODEL;
    _Float16* xi_ln = ws16 + off; off += NROW;    // LN(input_cap)  f16
    _Float16* xo_ln = ws16 + off; off += NROW;    // LN(output_cap) f16
    _Float16* wq16  = ws16 + off; off += NW;
    _Float16* wk16  = ws16 + off; off += NW;
    _Float16* wv16  = ws16 + off; off += NW;
    _Float16* wp16  = ws16 + off; off += NW;
    _Float16* q_buf = ws16 + off; off += NROW;    // [b,h,4096,64]
    _Float16* k_buf = ws16 + off; off += NROW;    // [b,h,4096,64]
    _Float16* vt_buf= ws16 + off; off += NROW;    // [b,h,64,4096]
    _Float16* ctx_b = ws16 + off; off += NROW;    // [b,4096,768]

    // 1) LayerNorm both inputs -> f16
    ln_f16_kernel<<<ROWS, 256, 0, stream>>>(input_cap,  ln_g, ln_b, xi_ln);
    ln_f16_kernel<<<ROWS, 256, 0, stream>>>(output_cap, ln_g, ln_b, xo_ln);

    // 2) Weights -> f16
    const int nw = D_MODEL * D_MODEL;
    cvt_f32_f16<<<nw / 256, 256, 0, stream>>>(Wq, wq16, nw);
    cvt_f32_f16<<<nw / 256, 256, 0, stream>>>(Wk, wk16, nw);
    cvt_f32_f16<<<nw / 256, 256, 0, stream>>>(Wv, wv16, nw);
    cvt_f32_f16<<<nw / 256, 256, 0, stream>>>(Wp, wp16, nw);

    // 3) Q/K/V projections (NT WMMA GEMM)
    dim3 ggrid(ROWS / 128, D_MODEL / 64);
    gemm_nt_kernel<<<ggrid, 256, 0, stream>>>(xo_ln, wq16, bq, q_buf,  nullptr, 0);
    gemm_nt_kernel<<<ggrid, 256, 0, stream>>>(xi_ln, wk16, bk, k_buf,  nullptr, 0);
    gemm_nt_kernel<<<ggrid, 256, 0, stream>>>(xi_ln, wv16, bv, vt_buf, nullptr, 1);

    // 4) Flash attention: 2*12*256 query tiles / 8 waves per block
    attn_kernel<<<768, 256, 0, stream>>>(q_buf, k_buf, vt_buf, ctx_b);

    // 5) Output projection + bias + residual -> f32 d_out
    gemm_nt_kernel<<<ggrid, 256, 0, stream>>>(ctx_b, wp16, bp, d_out, output_cap, 2);
}
